// CausalSelfAttention_24180665877196
// MI455X (gfx1250) — matrix operand order repaired and checked by reading, not compiled, not run on hardware
//
#include <hip/hip_runtime.h>

// ---------------------------------------------------------------------------
// CDNA5 (gfx1250) causal self-attention: bf16 WMMA everywhere, fp32 accum.
// v4: GEMM k-loop unrolled 2x (static ping-pong LDS buffers, alternating A
//     register sets -> no copy/nop rotation), and B tiles staged via the
//     Tensor Data Mover (tensor_load_to_lds + s_wait_tensorcnt) when the
//     builtin is available (cooperative ds_store staging as fallback).
// ---------------------------------------------------------------------------

typedef __bf16 bf16_t;
typedef bf16_t v16bf __attribute__((ext_vector_type(16)));
typedef bf16_t v8bf  __attribute__((ext_vector_type(8)));
typedef float  v8f   __attribute__((ext_vector_type(8)));
typedef unsigned int u32x4 __attribute__((ext_vector_type(4)));
typedef int i32x4 __attribute__((ext_vector_type(4)));
typedef int i32x8 __attribute__((ext_vector_type(8)));

#define T_LEN  4096
#define C_DIM  2048
#define N_HEAD 16
#define N_KV   4
#define HEAD_D 128
#define QD (N_HEAD * HEAD_D)   // 2048
#define KD (N_KV * HEAD_D)     // 512

#if defined(__gfx1250__) && __has_builtin(__builtin_amdgcn_tensor_load_to_lds)
#define USE_TDM 1
#else
#define USE_TDM 0
#endif

__device__ __forceinline__ bf16_t f2bf(float f) {
  unsigned int u = __builtin_bit_cast(unsigned int, f);
  u += 0x7FFFu + ((u >> 16) & 1u);            // round-to-nearest-even
  unsigned short s = (unsigned short)(u >> 16);
  return __builtin_bit_cast(bf16_t, s);
}

__device__ __forceinline__ v16bf cat16(v8bf lo, v8bf hi) {
  return __builtin_shufflevector(lo, hi, 0,1,2,3,4,5,6,7,8,9,10,11,12,13,14,15);
}

__device__ __forceinline__ v8f zero8() {
  v8f z;
#pragma unroll
  for (int i = 0; i < 8; ++i) z[i] = 0.0f;
  return z;
}

#if USE_TDM
// DMA a 64-row x 64-byte 2D tile (rows stride K bf16 elems) from global into
// LDS at byte offset lds_off, via the Tensor Data Mover. data_size = 8 bytes;
// tile = 8 units x 64 rows; dense in LDS (4KB).
__device__ __forceinline__ void tdm_load_tile(const bf16_t* gsrc,
                                              unsigned lds_off, int K) {
  unsigned long long ga = (unsigned long long)(size_t)gsrc;
  u32x4 g0;
  g0[0] = 1u;                                   // count=1, user descriptor
  g0[1] = lds_off;                              // lds_addr (bytes)
  g0[2] = (unsigned)ga;                         // global_addr[31:0]
  g0[3] = (unsigned)(ga >> 32) | (2u << 30);    // global_addr[56:32] | type=2
  i32x8 g1;
  g1[0] = (3 << 16);          // workgroup_mask=0, data_size=8B
  g1[1] = (8 << 16);          // tensor_dim0 = 8 units (in [79:48] low half)
  g1[2] = (64 << 16);         // tensor_dim0 hi=0 | tensor_dim1 = 64
  g1[3] = (8 << 16);          // tensor_dim1 hi=0 | tile_dim0 = 8 units
  g1[4] = 64;                 // tile_dim1 = 64, tile_dim2 = 0
  g1[5] = K >> 2;             // tensor_dim0_stride = K*2/8 units
  g1[6] = 0;
  g1[7] = 0;                  // tensor_dim1_stride = 0 (2D)
  i32x4 z4 = {0, 0, 0, 0};
#if __has_include(<hip/amd_detail/amd_gfx1250_TDM.h>)
  i32x8 z8 = {0, 0, 0, 0, 0, 0, 0, 0};
  __builtin_amdgcn_tensor_load_to_lds(g0, g1, z4, z4, z8, 0);
#else
  __builtin_amdgcn_tensor_load_to_lds(g0, g1, z4, z4, 0);
#endif
}
#endif

// --------------------------- conversion kernels ----------------------------

__global__ __launch_bounds__(256) void cvt_bf16(const float* __restrict__ in,
                                                bf16_t* __restrict__ out, int n) {
  int i = blockIdx.x * blockDim.x + threadIdx.x;
  if (i < n) out[i] = f2bf(in[i]);
}

// in: rows x cols (row-major fp32), out: cols x rows (row-major bf16) == in^T
__global__ __launch_bounds__(256) void cvt_bf16_T(const float* __restrict__ in,
                                                  bf16_t* __restrict__ out,
                                                  int rows, int cols) {
  int i = blockIdx.x * blockDim.x + threadIdx.x;
  if (i >= rows * cols) return;
  int r = i / cols, c = i % cols;
  out[(size_t)c * rows + r] = f2bf(in[i]);
}

// vf: (T, KV*D) fp32 -> vt: (KV, D, T) bf16
__global__ __launch_bounds__(256) void v_transpose(const float* __restrict__ vf,
                                                   bf16_t* __restrict__ vt) {
  int idx = blockIdx.x * blockDim.x + threadIdx.x;   // over KV*D*T
  int t   = idx % T_LEN;
  int rem = idx / T_LEN;
  int d   = rem % HEAD_D;
  int g   = rem / HEAD_D;
  vt[idx] = f2bf(vf[(size_t)t * KD + g * HEAD_D + d]);
}

// ------------------------------- WMMA GEMM ---------------------------------
// C[M x N] (f32) = A[M x K] (bf16, row-major) * B, with B supplied transposed
// as BT[N x K] (bf16, row-major).
// Block = 256 threads = 8 waves. Macro tile: 256 rows x 64 cols; each wave
// owns a 32x64 slice (two 16-row A subtiles sharing the same B fragments).
// k-loop unrolled 2x: stage A computes on bsh[0]/aA while bsh[1]/aB prefetch,
// stage B the reverse. B tiles move via TDM (wave 0) or cooperative staging.
__global__ __launch_bounds__(256) void gemm_bf16_f32(
    const bf16_t* __restrict__ A, const bf16_t* __restrict__ BT,
    float* __restrict__ C, int M, int N, int K) {
  __shared__ bf16_t bsh[2][64 * 32];          // 2 x 4KB B tiles (contiguous)
  int tilesN = N >> 6;
  int bm = (int)blockIdx.x / tilesN;
  int n0 = ((int)blockIdx.x % tilesN) << 6;
  int wslot = threadIdx.x >> 5;
  int lane  = threadIdx.x & 31;
  int m0    = (bm << 8) + (wslot << 5);       // this wave's 32 rows
  int col   = lane & 15;
  int half  = lane >> 4;

  v8f acc[2][4];
#pragma unroll
  for (int s = 0; s < 2; ++s)
#pragma unroll
    for (int t = 0; t < 4; ++t) acc[s][t] = zero8();

  const bf16_t* arow0 = A + (size_t)(m0 + col) * K;
  const bf16_t* arow1 = A + (size_t)(m0 + 16 + col) * K;
  const bf16_t* btile = BT + (size_t)n0 * K;  // row n0, this block's B panel

#if !USE_TDM
  // cooperative staging role: each thread moves one 16B segment of 4KB tile
  int tid  = threadIdx.x;
  int srow = tid >> 2;                        // 0..63  (column index n-n0)
  int sseg = tid & 3;                         // 0..3   (8-element k segment)
  const bf16_t* sgl = btile + (size_t)srow * K + sseg * 8;
  bf16_t* sd0 = &bsh[0][srow * 32 + sseg * 8];
  bf16_t* sd1 = &bsh[1][srow * 32 + sseg * 8];
#else
  unsigned lds0 = (unsigned)(size_t)&bsh[0][0];
  unsigned lds1 = lds0 + 64 * 32 * 2;
#endif

  // per-wave helpers -------------------------------------------------------
  auto load_a = [&](int k, v16bf (&a)[2]) {
    a[0] = cat16(*(const v8bf*)(arow0 + k + half * 8),
                 *(const v8bf*)(arow0 + k + 16 + half * 8));
    a[1] = cat16(*(const v8bf*)(arow1 + k + half * 8),
                 *(const v8bf*)(arow1 + k + 16 + half * 8));
  };
  auto compute = [&](const bf16_t* bcur, const v16bf (&a)[2]) {
    v16bf b[4];
#pragma unroll
    for (int t = 0; t < 4; ++t) {
      const bf16_t* brow = bcur + (t * 16 + col) * 32 + half * 8;
      b[t] = cat16(*(const v8bf*)brow, *(const v8bf*)(brow + 16));
    }
#pragma unroll
    for (int t = 0; t < 4; ++t)
      acc[0][t] = __builtin_amdgcn_wmma_f32_16x16x32_bf16(
          false, a[0], false, b[t], (short)0, acc[0][t], false, false);
#pragma unroll
    for (int t = 0; t < 4; ++t)
      acc[1][t] = __builtin_amdgcn_wmma_f32_16x16x32_bf16(
          false, a[1], false, b[t], (short)0, acc[1][t], false, false);
  };

  // prologue: stage k=0 into buffer 0; preload A fragments for k=0
  v16bf aA[2], aB[2];
  load_a(0, aA);
#if USE_TDM
  if (wslot == 0) tdm_load_tile(btile, lds0, K);
#else
  *(v8bf*)sd0 = *(const v8bf*)sgl;
#endif

  for (int k0 = 0; k0 < K; k0 += 64) {
    // ---------------- stage A: compute k0 on bsh[0]/aA ----------------
    int kn = k0 + 32;                          // always < K (K % 64 == 0)
#if USE_TDM
    if (wslot == 0) {
      tdm_load_tile(btile + kn, lds1, K);      // prefetch next tile -> buf 1
      __builtin_amdgcn_s_wait_tensorcnt(1);    // current tile (buf 0) ready
    }
    load_a(kn, aB);
    __syncthreads();
    compute(&bsh[0][0], aA);
    __syncthreads();                           // buf-0 reads done before reuse
#else
    v8bf nreg = *(const v8bf*)(sgl + kn);
    load_a(kn, aB);
    __syncthreads();
    *(v8bf*)sd1 = nreg;
    compute(&bsh[0][0], aA);
#endif
    // ---------------- stage B: compute kn on bsh[1]/aB ----------------
    int kn2 = k0 + 64;
    int hn  = kn2 < K;
    int kw  = hn ? kn2 : 0;
#if USE_TDM
    if (wslot == 0) {
      if (hn) {
        tdm_load_tile(btile + kw, lds0, K);    // prefetch -> buf 0
        __builtin_amdgcn_s_wait_tensorcnt(1);
      } else {
        __builtin_amdgcn_s_wait_tensorcnt(0);
      }
    }
    if (hn) load_a(kw, aA);
    __syncthreads();
    compute(&bsh[1][0], aB);
    __syncthreads();
#else
    v8bf nreg2 = *(const v8bf*)(sgl + kw);
    if (hn) load_a(kw, aA);
    __syncthreads();
    if (hn) *(v8bf*)sd0 = nreg2;
    compute(&bsh[1][0], aB);
#endif
  }

#pragma unroll
  for (int s = 0; s < 2; ++s) {
    int mb = m0 + s * 16 + half * 8;
#pragma unroll
    for (int t = 0; t < 4; ++t)
#pragma unroll
      for (int r = 0; r < 8; ++r)
        C[(size_t)(mb + r) * N + n0 + t * 16 + col] = acc[s][t][r];
  }
}

// ------------------------- fused RMSNorm + RoPE ----------------------------
// in: (T, nh*128) fp32; out: (nh, T, 128) bf16.  One wave per (t, h) row.
__global__ __launch_bounds__(256) void rmsnorm_rope(const float* __restrict__ in,
                                                    bf16_t* __restrict__ out,
                                                    int nrows, int nh) {
  int wid  = (int)((blockIdx.x * blockDim.x + threadIdx.x) >> 5);
  int lane = threadIdx.x & 31;
  if (wid >= nrows) return;
  int t = wid / nh, h = wid % nh;
  const float* r = in + ((size_t)t * nh + h) * HEAD_D;
  float f0 = r[lane], f1 = r[lane + 32], f2 = r[lane + 64], f3 = r[lane + 96];
  float ss = f0 * f0 + f1 * f1 + f2 * f2 + f3 * f3;
#pragma unroll
  for (int m = 16; m; m >>= 1) ss += __shfl_xor(ss, m, 32);
  float rr = rsqrtf(ss * (1.0f / 128.0f) + 1e-6f);
  f0 *= rr; f1 *= rr; f2 *= rr; f3 *= rr;
  // RoPE: pair (d, d+64) shares freq 10000^(-d/64); lane holds d=lane, lane+32
  const float NLOG = -9.210340371976184f / 64.0f;     // -ln(10000)/64
  float inv0 = __expf((float)lane * NLOG);
  float inv1 = __expf((float)(lane + 32) * NLOG);
  float s0, c0, s1, c1;
  __sincosf((float)t * inv0, &s0, &c0);
  __sincosf((float)t * inv1, &s1, &c1);
  float o0 = f0 * c0 - f2 * s0;
  float o2 = f2 * c0 + f0 * s0;
  float o1 = f1 * c1 - f3 * s1;
  float o3 = f3 * c1 + f1 * s1;
  bf16_t* w = out + ((size_t)h * T_LEN + t) * HEAD_D;
  w[lane] = f2bf(o0); w[lane + 32] = f2bf(o1);
  w[lane + 64] = f2bf(o2); w[lane + 96] = f2bf(o3);
}

// --------------------------- flash attention -------------------------------
// qb: (H, T, 128) bf16 ; kb: (KV, T, 128) bf16 ; vtb: (KV, 128, T) bf16
// yb: (T, H*128) bf16.  One wave per (head, 16-row q block); online softmax.
// K fragments batched in groups of 4 (one wait, 4 chained WMMAs); V fragments
// prefetched before the softmax VALU work so the PV WMMAs don't stall.
__global__ __launch_bounds__(256) void attn_kernel(const bf16_t* __restrict__ qb,
                                                   const bf16_t* __restrict__ kb,
                                                   const bf16_t* __restrict__ vtb,
                                                   bf16_t* __restrict__ yb) {
  __shared__ bf16_t sh[8][16 * 32];
  int wslot = threadIdx.x >> 5;
  int wave  = blockIdx.x * 8 + wslot;
  int lane  = threadIdx.x & 31;
  int qblocks = T_LEN / 16;                 // 256
  int h  = wave / qblocks;
  int i0 = (wave % qblocks) * 16;
  int g  = h / (N_HEAD / N_KV);
  const bf16_t* Q  = qb  + (size_t)h * T_LEN * HEAD_D;
  const bf16_t* Kp = kb  + (size_t)g * T_LEN * HEAD_D;
  const bf16_t* VT = vtb + (size_t)g * HEAD_D * T_LEN;
  bf16_t* lds = sh[wslot];

  int col  = lane & 15;
  int half = lane >> 4;

  // Q A-fragments for all 4 K-steps over D=128, loaded once
  v16bf qreg[4];
  const bf16_t* qrow = Q + (size_t)(i0 + col) * HEAD_D;
#pragma unroll
  for (int kk = 0; kk < 4; ++kk)
    qreg[kk] = cat16(*(const v8bf*)(qrow + kk * 32 + half * 16),
                     *(const v8bf*)(qrow + kk * 32 + half * 16 + 8));

  v8f o[8];
#pragma unroll
  for (int dt = 0; dt < 8; ++dt) o[dt] = zero8();
  float mrow[8], lrow[8];
#pragma unroll
  for (int r = 0; r < 8; ++r) { mrow[r] = -1e30f; lrow[r] = 0.0f; }

  const float scale = 0.08838834764831845f;  // 1/sqrt(128)
  int jmax = i0 + 16;
  for (int j0 = 0; j0 < jmax; j0 += 32) {
    // ---- S = Q Kt : two 16x16 score tiles over this 32-wide kv block ----
    v8f s0 = zero8(), s1 = zero8();
    const bf16_t* krow0 = Kp + (size_t)(j0 + col) * HEAD_D;
    const bf16_t* krow1 = Kp + (size_t)(j0 + 16 + col) * HEAD_D;
    v16bf kf4[4];
#pragma unroll
    for (int kk = 0; kk < 4; ++kk)
      kf4[kk] = cat16(*(const v8bf*)(krow0 + kk * 32 + half * 16),
                      *(const v8bf*)(krow0 + kk * 32 + half * 16 + 8));
#pragma unroll
    for (int kk = 0; kk < 4; ++kk)
      s0 = __builtin_amdgcn_wmma_f32_16x16x32_bf16(
          false, qreg[kk], false, kf4[kk], (short)0, s0, false, false);
#pragma unroll
    for (int kk = 0; kk < 4; ++kk)
      kf4[kk] = cat16(*(const v8bf*)(krow1 + kk * 32 + half * 16),
                      *(const v8bf*)(krow1 + kk * 32 + half * 16 + 8));
#pragma unroll
    for (int kk = 0; kk < 4; ++kk)
      s1 = __builtin_amdgcn_wmma_f32_16x16x32_bf16(
          false, qreg[kk], false, kf4[kk], (short)0, s1, false, false);
    // ---- prefetch V fragments: overlap with the softmax VALU below ----
    v16bf bv[8];
#pragma unroll
    for (int dt = 0; dt < 8; ++dt) {
      const bf16_t* vrow = VT + (size_t)(dt * 16 + col) * T_LEN + j0 + half * 8;
      bv[dt] = cat16(*(const v8bf*)vrow, *(const v8bf*)(vrow + 16));
    }
    // ---- causal mask, scale, row-max ----
    int qi0 = i0 + half * 8;
    int kja = j0 + col, kjb = kja + 16;
    float rmax[8];
#pragma unroll
    for (int r = 0; r < 8; ++r) {
      int qi = qi0 + r;
      float a = (kja > qi) ? -1e30f : s0[r] * scale;
      float b = (kjb > qi) ? -1e30f : s1[r] * scale;
      s0[r] = a; s1[r] = b;
      rmax[r] = fmaxf(a, b);
    }
#pragma unroll
    for (int m = 1; m < 16; m <<= 1)
#pragma unroll
      for (int r = 0; r < 8; ++r)
        rmax[r] = fmaxf(rmax[r], __shfl_xor(rmax[r], m, 32));
    // ---- online softmax update ----
    float alpha[8], psum[8];
#pragma unroll
    for (int r = 0; r < 8; ++r) {
      float nm = fmaxf(mrow[r], rmax[r]);
      alpha[r] = __expf(mrow[r] - nm);
      mrow[r]  = nm;
      float p0 = __expf(s0[r] - nm);
      float p1 = __expf(s1[r] - nm);
      psum[r] = p0 + p1;
      lds[(half * 8 + r) * 32 + col]      = f2bf(p0);
      lds[(half * 8 + r) * 32 + 16 + col] = f2bf(p1);
    }
#pragma unroll
    for (int m = 1; m < 16; m <<= 1)
#pragma unroll
      for (int r = 0; r < 8; ++r) psum[r] += __shfl_xor(psum[r], m, 32);
#pragma unroll
    for (int r = 0; r < 8; ++r) lrow[r] = lrow[r] * alpha[r] + psum[r];
#pragma unroll
    for (int dt = 0; dt < 8; ++dt)
#pragma unroll
      for (int r = 0; r < 8; ++r) o[dt][r] *= alpha[r];
    // ---- O += P V : P re-read from LDS in A-fragment layout ----
    v16bf pa = cat16(*(const v8bf*)(lds + col * 32 + half * 8),
                     *(const v8bf*)(lds + col * 32 + 16 + half * 8));
#pragma unroll
    for (int dt = 0; dt < 8; ++dt)
      o[dt] = __builtin_amdgcn_wmma_f32_16x16x32_bf16(
          false, pa, false, bv[dt], (short)0, o[dt], false, false);
  }
  // ---- normalize and write y (T, H*D) bf16 ----
#pragma unroll
  for (int dt = 0; dt < 8; ++dt)
#pragma unroll
    for (int r = 0; r < 8; ++r) {
      float val = o[dt][r] / lrow[r];
      yb[(size_t)(i0 + half * 8 + r) * QD + h * HEAD_D + dt * 16 + col] = f2bf(val);
    }
}

// ------------------------------ launcher -----------------------------------

extern "C" void kernel_launch(void* const* d_in, const int* in_sizes, int n_in,
                              void* d_out, int out_size, void* d_ws, size_t ws_size,
                              hipStream_t stream) {
  (void)in_sizes; (void)n_in; (void)out_size; (void)ws_size;
  const float* x  = (const float*)d_in[0];
  const float* Wq = (const float*)d_in[1];
  const float* Wk = (const float*)d_in[2];
  const float* Wv = (const float*)d_in[3];
  const float* Wo = (const float*)d_in[4];
  float* out = (float*)d_out;

  char* base = (char*)d_ws;
  size_t off = 0;
  auto alloc = [&](size_t bytes) -> char* {
    char* p = base + off;
    off += (bytes + 255) & ~(size_t)255;
    return p;
  };
  bf16_t* xb  = (bf16_t*)alloc((size_t)T_LEN * C_DIM * 2);
  bf16_t* wqT = (bf16_t*)alloc((size_t)QD * C_DIM * 2);
  bf16_t* wkT = (bf16_t*)alloc((size_t)KD * C_DIM * 2);
  bf16_t* wvT = (bf16_t*)alloc((size_t)KD * C_DIM * 2);
  bf16_t* woT = (bf16_t*)alloc((size_t)C_DIM * QD * 2);
  float*  qf  = (float*)alloc((size_t)T_LEN * QD * 4);
  float*  kf  = (float*)alloc((size_t)T_LEN * KD * 4);
  float*  vf  = (float*)alloc((size_t)T_LEN * KD * 4);
  bf16_t* qbf = (bf16_t*)alloc((size_t)N_HEAD * T_LEN * HEAD_D * 2);
  bf16_t* kbf = (bf16_t*)alloc((size_t)N_KV * T_LEN * HEAD_D * 2);
  bf16_t* vtb = (bf16_t*)alloc((size_t)N_KV * HEAD_D * T_LEN * 2);
  bf16_t* ybf = (bf16_t*)alloc((size_t)T_LEN * QD * 2);

  dim3 B256(256);
  // 1. precision conversion (+ weight transpose for WMMA B fragments)
  cvt_bf16<<<(T_LEN * C_DIM) / 256, B256, 0, stream>>>(x, xb, T_LEN * C_DIM);
  cvt_bf16_T<<<(C_DIM * QD) / 256, B256, 0, stream>>>(Wq, wqT, C_DIM, QD);
  cvt_bf16_T<<<(C_DIM * KD) / 256, B256, 0, stream>>>(Wk, wkT, C_DIM, KD);
  cvt_bf16_T<<<(C_DIM * KD) / 256, B256, 0, stream>>>(Wv, wvT, C_DIM, KD);
  cvt_bf16_T<<<(QD * C_DIM) / 256, B256, 0, stream>>>(Wo, woT, QD, C_DIM);
  // 2. QKV projections (WMMA bf16 -> f32); grid = (M/256) * (N/64)
  gemm_bf16_f32<<<(T_LEN / 256) * (QD / 64), B256, 0, stream>>>(xb, wqT, qf, T_LEN, QD, C_DIM);
  gemm_bf16_f32<<<(T_LEN / 256) * (KD / 64), B256, 0, stream>>>(xb, wkT, kf, T_LEN, KD, C_DIM);
  gemm_bf16_f32<<<(T_LEN / 256) * (KD / 64), B256, 0, stream>>>(xb, wvT, vf, T_LEN, KD, C_DIM);
  // 3. RMSNorm + RoPE (q, k) ; V transpose to (KV, D, T)
  rmsnorm_rope<<<(T_LEN * N_HEAD) / 8, B256, 0, stream>>>(qf, qbf, T_LEN * N_HEAD, N_HEAD);
  rmsnorm_rope<<<(T_LEN * N_KV) / 8, B256, 0, stream>>>(kf, kbf, T_LEN * N_KV, N_KV);
  v_transpose<<<(N_KV * HEAD_D * T_LEN) / 256, B256, 0, stream>>>(vf, vtb);
  // 4. causal GQA flash attention (WMMA)
  attn_kernel<<<(N_HEAD * (T_LEN / 16)) / 8, B256, 0, stream>>>(qbf, kbf, vtb, ybf);
  // 5. output projection (WMMA) -> f32 d_out
  gemm_bf16_f32<<<(T_LEN / 256) * (C_DIM / 64), B256, 0, stream>>>(ybf, woT, out, T_LEN, C_DIM, QD);
}
